// GatedDeltaNetEncoder_90142773609389
// MI455X (gfx1250) — matrix operand order repaired and checked by reading, not compiled, not run on hardware
//
#include <hip/hip_runtime.h>
#include <hip/hip_bf16.h>

// ---------------------------------------------------------------------------
// Types for CDNA5 WMMA (wave32) + TDM descriptors
// ---------------------------------------------------------------------------
typedef __bf16 bf16_t;
typedef bf16_t v16bf __attribute__((ext_vector_type(16)));
typedef float  v8f   __attribute__((ext_vector_type(8)));
typedef unsigned int u32x4 __attribute__((ext_vector_type(4)));
typedef int          i32x4 __attribute__((ext_vector_type(4)));
typedef int          i32x8 __attribute__((ext_vector_type(8)));

#define BM 128
#define BN 128
#define BK 32
#define FSTRIDE 36                  // f32 LDS row stride: 32 data + 4 pad DWORDs (TDM pad)
#define TILE_F32 (BM * FSTRIDE)     // floats per tile buffer (4608)
#define GEMM_LDS_BYTES (4 * TILE_F32 * sizeof(float))   // A[2] + B[2] buffers

// ---------------------------------------------------------------------------
// Issue one TDM tensor_load_to_lds for a (rows x BK) f32 tile.
// D# per CDNA5 ISA ch.8: group0 = {flags, lds_addr, global_addr, type=2},
// group1 = {data_size=4B, pad_enable, pad_interval=32dw, pad_amount=4dw,
//           tensor_dim0=lda, tensor_dim1=rows, tile_dim0=32, tile_dim1=128,
//           tensor_dim0_stride=lda}. Rows beyond tensor_dim1 return zero.
// This toolchain exposes the 6-arg builtin (4 descriptor groups + spare + cpol).
// ---------------------------------------------------------------------------
__device__ __forceinline__ void tdm_tile_load(const float* gsrc, unsigned lds_off,
                                              int rows, int lda)
{
  unsigned long long ga = (unsigned long long)(uintptr_t)gsrc;
  u32x4 g0;
  g0[0] = 1u;                                            // count=1, user descriptor
  g0[1] = lds_off;                                       // lds_addr (bytes)
  g0[2] = (unsigned)ga;                                  // global_addr[31:0]
  g0[3] = (unsigned)((ga >> 32) & 0x01ffffffu) | (2u << 30); // addr[56:32] | type=2
  unsigned td0 = (unsigned)lda;                          // tensor_dim0 (elements)
  unsigned td1 = (unsigned)rows;                         // tensor_dim1 (rows)
  i32x8 g1;
  g1[0] = (int)((2u << 16) | (1u << 20) | (4u << 22) | (3u << 25));
  g1[1] = (int)((td0 & 0xffffu) << 16);                  // [15:0] atomic addr=0, [31:16] td0 lo
  g1[2] = (int)((td0 >> 16) | ((td1 & 0xffffu) << 16));  // td0 hi | td1 lo
  g1[3] = (int)((td1 >> 16) | ((unsigned)BK << 16));     // td1 hi | tile_dim0=32
  g1[4] = (int)(unsigned)BM;                             // tile_dim1=128, tile_dim2=0
  g1[5] = (int)td0;                                      // tensor_dim0_stride lo32
  g1[6] = 0;                                             // stride0 hi | stride1 lo
  g1[7] = 0;
  i32x4 z4 = {0, 0, 0, 0};
  i32x8 z8 = {0, 0, 0, 0, 0, 0, 0, 0};
  __builtin_amdgcn_tensor_load_to_lds(g0, g1, z4, z4, z8, 0);
}

// build a v16bf fragment from two 8-float LDS runs (f32 -> bf16 convert).
// Explicit float4 reads force ds_load_b128 (all runs are 16B-aligned:
// row stride 144B, run offsets 0/32/64/96B within the row).
__device__ __forceinline__ v16bf frag_from_f32(const float* p0, const float* p1)
{
  float4 a = *reinterpret_cast<const float4*>(p0);
  float4 b = *reinterpret_cast<const float4*>(p0 + 4);
  float4 c = *reinterpret_cast<const float4*>(p1);
  float4 d = *reinterpret_cast<const float4*>(p1 + 4);
  v16bf f;
  f[0]  = (bf16_t)a.x; f[1]  = (bf16_t)a.y; f[2]  = (bf16_t)a.z; f[3]  = (bf16_t)a.w;
  f[4]  = (bf16_t)b.x; f[5]  = (bf16_t)b.y; f[6]  = (bf16_t)b.z; f[7]  = (bf16_t)b.w;
  f[8]  = (bf16_t)c.x; f[9]  = (bf16_t)c.y; f[10] = (bf16_t)c.z; f[11] = (bf16_t)c.w;
  f[12] = (bf16_t)d.x; f[13] = (bf16_t)d.y; f[14] = (bf16_t)d.z; f[15] = (bf16_t)d.w;
  return f;
}

// ---------------------------------------------------------------------------
// Tiled bf16 WMMA GEMM:  C[M,N] = A[M,K] @ W[N,K]^T  (+ optional residual)
// TDM double-buffered staging (f32 tiles in LDS, padded rows), one wave
// drives the DMA; 8 waves each run 4x2 v_wmma_f32_16x16x32_bf16 per K=32.
// ---------------------------------------------------------------------------
template<bool GUARD>
__global__ __launch_bounds__(256)
void gemm_bf16_wmma(const float* __restrict__ A, int lda,
                    const float* __restrict__ W, int ldw,
                    float* __restrict__ C, int ldc,
                    const float* __restrict__ resid,
                    int M, int N, int K)
{
  extern __shared__ float smem[];
  float* Af = smem;                    // [2][TILE_F32]
  float* Bf = smem + 2 * TILE_F32;     // [2][TILE_F32]

  const int tid  = threadIdx.x;
  const int lane = tid & 31;
  const int wave = tid >> 5;
  const int wm   = wave >> 2;          // 0..1 : 64 rows
  const int wn   = wave & 3;           // 0..3 : 32 cols
  const int m0   = blockIdx.y * BM;
  const int n0   = blockIdx.x * BN;
  const int l16  = lane & 15;
  const int half = lane >> 4;
  const int kbA  = half * 8;
  const int kbB  = half * 16;

  const unsigned ldsA0 = (unsigned)(uintptr_t)(void*)Af;
  const unsigned ldsB0 = (unsigned)(uintptr_t)(void*)Bf;
  const int rowsA = GUARD ? ((M - m0) < BM ? (M - m0) : BM) : BM;
  const int rowsB = GUARD ? ((N - n0) < BN ? (N - n0) : BN) : BN;

  v8f zero = {};
  v8f acc[4][2];
#pragma unroll
  for (int i = 0; i < 4; ++i)
#pragma unroll
    for (int j = 0; j < 2; ++j) acc[i][j] = zero;

  const int NT = K / BK;               // K is always a multiple of 32 here

  if (tid == 0) {                      // wave 0 drives the TDM
    tdm_tile_load(A + (size_t)m0 * lda, ldsA0, rowsA, lda);
    tdm_tile_load(W + (size_t)n0 * ldw, ldsB0, rowsB, ldw);
  }

  for (int i = 0; i < NT; ++i) {
    if (tid == 0) __builtin_amdgcn_s_wait_tensorcnt(0);
    __syncthreads();                   // tile i resident in LDS for all waves

    if (tid == 0 && (i + 1) < NT) {    // kick DMA for tile i+1 into other buffer
      int nb = (i + 1) & 1;
      tdm_tile_load(A + (size_t)m0 * lda + (size_t)(i + 1) * BK,
                    ldsA0 + (unsigned)(nb * TILE_F32 * 4), rowsA, lda);
      tdm_tile_load(W + (size_t)n0 * ldw + (size_t)(i + 1) * BK,
                    ldsB0 + (unsigned)(nb * TILE_F32 * 4), rowsB, ldw);
    }

    const int cb = i & 1;
    const float* Ab = Af + cb * TILE_F32;
    const float* Bb = Bf + cb * TILE_F32;

    v16bf afrag[4], bfrag[2];
#pragma unroll
    for (int mi = 0; mi < 4; ++mi) {
      const float* p = Ab + (wm * 64 + mi * 16 + l16) * FSTRIDE;
      afrag[mi] = frag_from_f32(p + kbA, p + 16 + kbA);
    }
#pragma unroll
    for (int ni = 0; ni < 2; ++ni) {
      const float* p = Bb + (wn * 32 + ni * 16 + l16) * FSTRIDE + kbB;
      bfrag[ni] = frag_from_f32(p - kbB + kbA, p - kbB + 16 + kbA);
    }
#pragma unroll
    for (int mi = 0; mi < 4; ++mi)
#pragma unroll
      for (int ni = 0; ni < 2; ++ni)
        acc[mi][ni] = __builtin_amdgcn_wmma_f32_16x16x32_bf16(
            false, afrag[mi], false, bfrag[ni],
            (short)0, acc[mi][ni], false, false);
    __syncthreads();                   // done reading buffer cb before its reuse
  }

  // ---- epilogue: C/D layout: element e -> row = half*8+e, col = l16 ----
#pragma unroll
  for (int mi = 0; mi < 4; ++mi)
#pragma unroll
    for (int ni = 0; ni < 2; ++ni)
#pragma unroll
      for (int e = 0; e < 8; ++e) {
        int r = m0 + wm * 64 + mi * 16 + half * 8 + e;
        int c = n0 + wn * 32 + ni * 16 + l16;
        if (!GUARD || (r < M && c < N)) {
          float v = acc[mi][ni][e];
          if (resid) v += resid[(size_t)r * ldc + c];
          C[(size_t)r * ldc + c] = v;
        }
      }
}

// ---------------------------------------------------------------------------
// RMSNorm over rows of length L: y = x * rsqrt(mean(x^2)+eps) * w
// ---------------------------------------------------------------------------
__global__ void rms_rows(const float* __restrict__ X, const float* __restrict__ w,
                         float* __restrict__ Y, int L)
{
  int row = blockIdx.x;
  const float* x = X + (size_t)row * L;
  float* y = Y + (size_t)row * L;
  float ss = 0.f;
  for (int i = threadIdx.x; i < L; i += blockDim.x) { float v = x[i]; ss += v * v; }
  for (int off = 16; off; off >>= 1) ss += __shfl_xor(ss, off, 32);
  __shared__ float part[8];
  if ((threadIdx.x & 31) == 0) part[threadIdx.x >> 5] = ss;
  __syncthreads();
  float tot = 0.f;
  for (int i = 0; i < (int)(blockDim.x >> 5); ++i) tot += part[i];
  float r = rsqrtf(tot / (float)L + 1e-6f);
  for (int i = threadIdx.x; i < L; i += blockDim.x) y[i] = x[i] * r * w[i];
}

// L2 normalize rows in place (sum, not mean)
__global__ void l2norm_rows(float* __restrict__ X, int L)
{
  int row = blockIdx.x;
  float* x = X + (size_t)row * L;
  float ss = 0.f;
  for (int i = threadIdx.x; i < L; i += blockDim.x) { float v = x[i]; ss += v * v; }
  for (int off = 16; off; off >>= 1) ss += __shfl_xor(ss, off, 32);
  __shared__ float part[8];
  if ((threadIdx.x & 31) == 0) part[threadIdx.x >> 5] = ss;
  __syncthreads();
  float tot = 0.f;
  for (int i = 0; i < (int)(blockDim.x >> 5); ++i) tot += part[i];
  float r = rsqrtf(tot + 1e-6f);
  for (int i = threadIdx.x; i < L; i += blockDim.x) x[i] *= r;
}

// ---------------------------------------------------------------------------
// State fusion: gate = sigmoid(G1 + G2[b] + bias); xf = gate*x + (1-gate)*es[b]
// ---------------------------------------------------------------------------
__global__ void fuse_state(const float* __restrict__ x, const float* __restrict__ G1,
                           const float* __restrict__ G2, const float* __restrict__ es,
                           const float* __restrict__ bias, float* __restrict__ xf,
                           int n, int TD, int Dm)
{
  int i = blockIdx.x * blockDim.x + threadIdx.x;
  if (i >= n) return;
  int b = i / TD;
  int d = i % Dm;
  float g = G1[i] + G2[b * Dm + d] + bias[d];
  g = 1.f / (1.f + __expf(-g));
  float e = es[b * Dm + d];
  xf[i] = g * x[i] + (1.f - g) * e;
}

// ---------------------------------------------------------------------------
// Causal depthwise conv (K=4) + SiLU, in place; window carried in registers,
// streaming prefetch ahead of the sequential walk.
// ---------------------------------------------------------------------------
__global__ void dwconv_silu(float* __restrict__ X, const float* __restrict__ Wc,
                            int Bn, int Tn, int Cn)
{
  int id = blockIdx.x * blockDim.x + threadIdx.x;
  if (id >= Bn * Cn) return;
  int b = id / Cn, c = id % Cn;
  float w0 = Wc[c * 4 + 0], w1 = Wc[c * 4 + 1], w2 = Wc[c * 4 + 2], w3 = Wc[c * 4 + 3];
  float z1 = 0.f, z2 = 0.f, z3 = 0.f;
  float* p = X + (size_t)b * Tn * Cn + c;
  for (int t = 0; t < Tn; ++t) {
    if (t + 16 < Tn) __builtin_prefetch(&p[(size_t)(t + 16) * Cn], 0, 0);
    float cur = p[(size_t)t * Cn];
    float y = w3 * cur + w2 * z1 + w1 * z2 + w0 * z3;
    y = y / (1.f + __expf(-y));
    p[(size_t)t * Cn] = y;
    z3 = z2; z2 = z1; z1 = cur;
  }
}

// g = -exp(A_log[h]) * softplus(aa + dt_bias[h]) (in place on aa)
// beta = 2*sigmoid(bb)                           (in place on bb)
__global__ void gate_beta(float* __restrict__ aa, float* __restrict__ bb,
                          const float* __restrict__ Alog, const float* __restrict__ dtb,
                          int n_g, int n_b, int Hn)
{
  int i = blockIdx.x * blockDim.x + threadIdx.x;
  if (i < n_g) {
    int h = i % Hn;
    float xv = aa[i] + dtb[h];
    float sp = (xv > 20.f) ? xv : log1pf(__expf(xv));
    aa[i] = -__expf(Alog[h]) * sp;
  }
  if (i < n_b) bb[i] = 2.f / (1.f + __expf(-bb[i]));
}

// ---------------------------------------------------------------------------
// Gated DeltaProduct recurrence. One workgroup per (b,h); state S[256][257]
// (padded) in dynamic LDS (~263 KB, fits CDNA5's 320 KB WGP LDS).
// Thread tid owns V-column tid -> no intra-token barriers on S.
// Decay folded into Householder i=0's update.
// ---------------------------------------------------------------------------
__global__ __launch_bounds__(256)
void delta_scan(const float* __restrict__ q,    // [B,T,H,256]
                const float* __restrict__ k,    // [B,T,2,H,256]
                const float* __restrict__ v,    // [B,T,2,H,256]
                const float* __restrict__ beta, // [B,T,2,H]
                const float* __restrict__ g,    // [B,T,H]
                float* __restrict__ o,          // [B,T,H,256]
                int Tn)
{
  extern __shared__ float smem[];
  const int DKc = 256, DVc = 256, SP = DVc + 1;
  float* S  = smem;               // DKc * SP
  float* kb = S + DKc * SP;       // 256
  float* vb = kb + DKc;           // 256
  float* qb = vb + DVc;           // 256

  const int b  = blockIdx.x >> 2;   // / H (H=4)
  const int h  = blockIdx.x & 3;
  const int vc = threadIdx.x;       // owned V column
  const float scale = 0.0625f;      // 256^-0.5

  for (int kk = 0; kk < DKc; ++kk) S[kk * SP + vc] = 0.f;
  __syncthreads();

  for (int t = 0; t < Tn; ++t) {
    size_t base = (size_t)b * Tn + t;
    float gd = __expf(g[base * 4 + h]);

    // ---- Householder 0 (decay folded in) ----
    kb[vc] = k[((base * 2 + 0) * 4 + h) * 256 + vc];
    vb[vc] = v[((base * 2 + 0) * 4 + h) * 256 + vc];
    __syncthreads();
    float bet = beta[(base * 2 + 0) * 4 + h];
    float pred = 0.f;
    for (int kk = 0; kk < DKc; ++kk) pred += kb[kk] * S[kk * SP + vc];
    pred *= gd;
    float u = (vb[vc] - pred) * bet;
    for (int kk = 0; kk < DKc; ++kk) {
      int idx = kk * SP + vc;
      S[idx] = S[idx] * gd + kb[kk] * u;
    }
    __syncthreads();

    // ---- Householder 1 ----
    kb[vc] = k[((base * 2 + 1) * 4 + h) * 256 + vc];
    vb[vc] = v[((base * 2 + 1) * 4 + h) * 256 + vc];
    qb[vc] = q[(base * 4 + h) * 256 + vc];
    __syncthreads();
    bet = beta[(base * 2 + 1) * 4 + h];
    pred = 0.f;
    for (int kk = 0; kk < DKc; ++kk) pred += kb[kk] * S[kk * SP + vc];
    u = (vb[vc] - pred) * bet;
    for (int kk = 0; kk < DKc; ++kk) {
      int idx = kk * SP + vc;
      S[idx] += kb[kk] * u;
    }
    // ---- query readout (post-update state, own column) ----
    float ov = 0.f;
    for (int kk = 0; kk < DKc; ++kk) ov += qb[kk] * S[kk * SP + vc];
    o[(base * 4 + h) * 256 + vc] = ov * scale;
    __syncthreads();
  }
}

// m = silu(gp) * up  (in place on gp)
__global__ void swiglu(float* __restrict__ gp, const float* __restrict__ up, size_t n)
{
  size_t i = (size_t)blockIdx.x * blockDim.x + threadIdx.x;
  if (i >= n) return;
  float a = gp[i];
  a = a / (1.f + __expf(-a));
  gp[i] = a * up[i];
}

__global__ void copy_last(const float* __restrict__ out, float* __restrict__ dst,
                          int Tn, int Dm)
{
  int i = blockIdx.x * blockDim.x + threadIdx.x;
  if (i >= 8 * Dm) return;
  int b = i / Dm, d = i % Dm;
  dst[i] = out[((size_t)b * Tn + (Tn - 1)) * Dm + d];
}

// ---------------------------------------------------------------------------
// Launch
// ---------------------------------------------------------------------------
extern "C" void kernel_launch(void* const* d_in, const int* in_sizes, int n_in,
                              void* d_out, int out_size, void* d_ws, size_t ws_size,
                              hipStream_t stream)
{
  (void)in_sizes; (void)n_in; (void)out_size; (void)ws_size;
  const int Bc = 8, Tc = 2048, Dc = 1024, Hc = 4, DKc = 256, DVc = 256, NHc = 2, Ic = 2816;
  const int MT = Bc * Tc;  // 16384

  const float* x    = (const float*)d_in[0];
  const float* prev = (const float*)d_in[1];
  const float* spw  = (const float*)d_in[2];
  const float* sgw  = (const float*)d_in[3];
  const float* sgb  = (const float*)d_in[4];
  const float* anw  = (const float*)d_in[5];
  const float* qw   = (const float*)d_in[6];
  const float* kw   = (const float*)d_in[7];
  const float* vw   = (const float*)d_in[8];
  const float* bw   = (const float*)d_in[9];
  const float* aw   = (const float*)d_in[10];
  const float* Alog = (const float*)d_in[11];
  const float* dtb  = (const float*)d_in[12];
  const float* qcw  = (const float*)d_in[13];
  const float* kcw  = (const float*)d_in[14];
  const float* vcw  = (const float*)d_in[15];
  const float* onw  = (const float*)d_in[16];
  const float* ow   = (const float*)d_in[17];
  const float* mnw  = (const float*)d_in[18];
  const float* gw   = (const float*)d_in[19];
  const float* uw   = (const float*)d_in[20];
  const float* dw   = (const float*)d_in[21];

  float* out      = (float*)d_out;
  float* out_last = out + (size_t)MT * Dc;

  char* cur = (char*)d_ws;
  auto alloc = [&](size_t nf) -> float* {
    float* p = (float*)cur;
    cur += ((nf * sizeof(float) + 255) & ~(size_t)255);
    return p;
  };
  float* es   = alloc((size_t)Bc * Dc);
  float* G2   = alloc((size_t)Bc * Dc);
  float* G1   = alloc((size_t)MT * Dc);
  float* xf   = alloc((size_t)MT * Dc);
  float* hbuf = alloc((size_t)MT * Dc);
  float* qp   = alloc((size_t)MT * Hc * DKc);
  float* kp   = alloc((size_t)MT * NHc * Hc * DKc);
  float* vp   = alloc((size_t)MT * NHc * Hc * DVc);
  float* bb   = alloc((size_t)MT * NHc * Hc);
  float* aa   = alloc((size_t)MT * Hc);
  float* oo   = alloc((size_t)MT * Hc * DVc);
  float* x2   = alloc((size_t)MT * Dc);
  float* gp   = alloc((size_t)MT * Ic);
  float* up   = alloc((size_t)MT * Ic);
  float* onrm = G1;  // G1 dead after fuse_state; reuse for head-normalized o

  dim3 blk(256);
  auto gg = [](int M, int N) {
    return dim3((unsigned)((N + 127) / 128), (unsigned)((M + 127) / 128));
  };
  const size_t GL = GEMM_LDS_BYTES;

  // ---- state fusion ----
  gemm_bf16_wmma<true><<<gg(Bc, Dc), blk, GL, stream>>>(prev, Dc, spw, Dc, es, Dc, nullptr, Bc, Dc, Dc);
  gemm_bf16_wmma<true><<<gg(Bc, Dc), blk, GL, stream>>>(es, Dc, sgw + Dc, 2 * Dc, G2, Dc, nullptr, Bc, Dc, Dc);
  gemm_bf16_wmma<false><<<gg(MT, Dc), blk, GL, stream>>>(x, Dc, sgw, 2 * Dc, G1, Dc, nullptr, MT, Dc, Dc);
  { int n = MT * Dc;
    fuse_state<<<(n + 255) / 256, blk, 0, stream>>>(x, G1, G2, es, sgb, xf, n, Tc * Dc, Dc); }

  // ---- attention projections ----
  rms_rows<<<MT, blk, 0, stream>>>(xf, anw, hbuf, Dc);
  gemm_bf16_wmma<false><<<gg(MT, Hc * DKc), blk, GL, stream>>>(hbuf, Dc, qw, Dc, qp, Hc * DKc, nullptr, MT, Hc * DKc, Dc);
  gemm_bf16_wmma<false><<<gg(MT, NHc * Hc * DKc), blk, GL, stream>>>(hbuf, Dc, kw, Dc, kp, NHc * Hc * DKc, nullptr, MT, NHc * Hc * DKc, Dc);
  gemm_bf16_wmma<false><<<gg(MT, NHc * Hc * DVc), blk, GL, stream>>>(hbuf, Dc, vw, Dc, vp, NHc * Hc * DVc, nullptr, MT, NHc * Hc * DVc, Dc);
  gemm_bf16_wmma<true><<<gg(MT, NHc * Hc), blk, GL, stream>>>(hbuf, Dc, bw, Dc, bb, NHc * Hc, nullptr, MT, NHc * Hc, Dc);
  gemm_bf16_wmma<true><<<gg(MT, Hc), blk, GL, stream>>>(hbuf, Dc, aw, Dc, aa, Hc, nullptr, MT, Hc, Dc);

  dwconv_silu<<<(Bc * Hc * DKc + 255) / 256, blk, 0, stream>>>(qp, qcw, Bc, Tc, Hc * DKc);
  dwconv_silu<<<(Bc * NHc * Hc * DKc + 255) / 256, blk, 0, stream>>>(kp, kcw, Bc, Tc, NHc * Hc * DKc);
  dwconv_silu<<<(Bc * NHc * Hc * DVc + 255) / 256, blk, 0, stream>>>(vp, vcw, Bc, Tc, NHc * Hc * DVc);

  l2norm_rows<<<MT * Hc, blk, 0, stream>>>(qp, DKc);
  l2norm_rows<<<MT * NHc * Hc, blk, 0, stream>>>(kp, DKc);

  gate_beta<<<(MT * NHc * Hc + 255) / 256, blk, 0, stream>>>(aa, bb, Alog, dtb, MT * Hc, MT * NHc * Hc, Hc);

  // ---- sequential delta-rule scan ----
  size_t smem = ((size_t)DKc * (DVc + 1) + 3 * 256) * sizeof(float);
  delta_scan<<<Bc * Hc, blk, smem, stream>>>(qp, kp, vp, bb, aa, oo, Tc);

  // ---- output proj (+residual) ----
  rms_rows<<<MT * Hc, blk, 0, stream>>>(oo, onw, onrm, DVc);
  gemm_bf16_wmma<false><<<gg(MT, Dc), blk, GL, stream>>>(onrm, Hc * DVc, ow, Hc * DVc, x2, Dc, xf, MT, Dc, Hc * DVc);

  // ---- gated MLP ----
  rms_rows<<<MT, blk, 0, stream>>>(x2, mnw, hbuf, Dc);
  gemm_bf16_wmma<false><<<gg(MT, Ic), blk, GL, stream>>>(hbuf, Dc, gw, Dc, gp, Ic, nullptr, MT, Ic, Dc);
  gemm_bf16_wmma<false><<<gg(MT, Ic), blk, GL, stream>>>(hbuf, Dc, uw, Dc, up, Ic, nullptr, MT, Ic, Dc);
  { size_t n = (size_t)MT * Ic;
    swiglu<<<(unsigned)((n + 255) / 256), blk, 0, stream>>>(gp, up, n); }
  gemm_bf16_wmma<false><<<gg(MT, Dc), blk, GL, stream>>>(gp, Ic, dw, Ic, out, Dc, x2, MT, Dc, Ic);

  copy_last<<<(Bc * Dc + 255) / 256, blk, 0, stream>>>(out, out_last, Tc, Dc);
}
